// SimpleGCN_39264591020170
// MI455X (gfx1250) — matrix operand order repaired and checked by reading, not compiled, not run on hardware
//
#include <hip/hip_runtime.h>
#include <hip/hip_bf16.h>

// ---- problem constants ----
#define NN   30000      // nodes
#define NE   120000     // edges
#define INF_ 9          // input features
#define EDF  3          // edge features
#define H    32         // hidden
#define GG   256        // graphs
#define KTOT 1056       // 1024 (outer-product) + 32 (bias matvec folded into K)
#define BROW 1064       // padded LDS row stride (halves) -> conflict-free ds_load_b128

typedef __attribute__((ext_vector_type(16))) _Float16 v16h;
typedef __attribute__((ext_vector_type(8)))  _Float16 v8h;
typedef __attribute__((ext_vector_type(8)))  float    v8f;

static __device__ inline v16h cat8(v8h a, v8h b) {
  return __builtin_shufflevector(a, b, 0,1,2,3,4,5,6,7,8,9,10,11,12,13,14,15);
}

// ---- h = relu(x @ Wn + bn); also emit f16 copy ----
__global__ void node_embed_kernel(const float* __restrict__ x, const float* __restrict__ Wn,
                                  const float* __restrict__ bn, float* __restrict__ hf,
                                  _Float16* __restrict__ h16) {
  int t = blockIdx.x * blockDim.x + threadIdx.x;
  if (t >= NN * H) return;
  int n = t >> 5, o = t & 31;
  float acc = bn[o];
#pragma unroll
  for (int i = 0; i < INF_; ++i) acc += x[n * INF_ + i] * Wn[i * H + o];
  acc = fmaxf(acc, 0.f);
  hf[t] = acc;
  h16[t] = (_Float16)acc;
}

// ---- ea = relu(edge_attr @ We + be) -> f16 ----
__global__ void edge_embed_kernel(const float* __restrict__ eattr, const float* __restrict__ We,
                                  const float* __restrict__ be, _Float16* __restrict__ ea16) {
  int t = blockIdx.x * blockDim.x + threadIdx.x;
  if (t >= NE * H) return;
  int e = t >> 5, o = t & 31;
  float acc = be[o];
#pragma unroll
  for (int j = 0; j < EDF; ++j) acc += eattr[e * EDF + j] * We[j * H + o];
  ea16[t] = (_Float16)fmaxf(acc, 0.f);
}

// ---- B' prep: Bp[n][i*32+j] = Wnn[k][j][i*32+n]; Bp[n][1024+i] = bnn[k][i*32+n] ----
__global__ void bprep_kernel(const float* __restrict__ Wnn, const float* __restrict__ bnn,
                             _Float16* __restrict__ Bp, int k) {
  int t = blockIdx.x * blockDim.x + threadIdx.x;
  if (t >= H * KTOT) return;
  int n = t / KTOT, kk = t % KTOT;
  float v;
  if (kk < 1024) {
    int i = kk >> 5, j = kk & 31;
    v = Wnn[(size_t)k * H * H * H + j * (H * H) + i * H + n];
  } else {
    int i = kk - 1024;
    v = bnn[k * (H * H) + i * H + n];
  }
  Bp[t] = (_Float16)v;
}

__global__ void zero_kernel(float* __restrict__ p, int n) {
  int t = blockIdx.x * blockDim.x + threadIdx.x;
  if (t < n) p[t] = 0.f;
}

// ---- the heavy kernel: per-edge msg via WMMA GEMM [16 edges,1056] x [1056,32], scatter-add ----
__global__ void __launch_bounds__(256) edge_msg_kernel(
    const _Float16* __restrict__ h16, const _Float16* __restrict__ ea16,
    const _Float16* __restrict__ Bp, const int* __restrict__ srcIdx,
    const int* __restrict__ dstIdx, float* __restrict__ agg, int ntiles) {
  extern __shared__ _Float16 Bs[];  // [32][BROW] halves

  const int tid = threadIdx.x;
  // cooperative load of B' (32 x 1056 f16) into padded LDS
  for (int c = tid; c < H * (KTOT / 8); c += blockDim.x) {
    int row = c / (KTOT / 8);
    int c8 = c % (KTOT / 8);
    *(v8h*)(&Bs[row * BROW + c8 * 8]) = *(const v8h*)(&Bp[row * KTOT + c8 * 8]);
  }
  __syncthreads();

  const int lane = tid & 31;
  const int lo = lane & 15;   // edge-in-tile (A rows / C cols)
  const int hb = lane >> 4;   // lane-group select for K slices
  const int wavesPerGrid = (gridDim.x * blockDim.x) >> 5;
  const int waveGlobal = (blockIdx.x * blockDim.x + tid) >> 5;

  const _Float16* brow0 = &Bs[(size_t)lo * BROW + 16 * hb];         // N = lo
  const _Float16* brow1 = &Bs[(size_t)(16 + lo) * BROW + 16 * hb];  // N = 16+lo

  for (int tile = waveGlobal; tile < ntiles; tile += wavesPerGrid) {
    // Stop LLVM from hoisting the (loop-invariant) LDS fragment loads out of
    // the tile loop: hoisting 66 fragments = 528 VGPRs -> scratch spills.
    asm volatile("" ::: "memory");

    const int e = tile * 16 + lo;
    const int src = srcIdx[e];
    const _Float16* hp = h16 + (size_t)src * H;
    const _Float16* ep = ea16 + (size_t)e * H;

    v16h hf0 = *(const v16h*)(hp);       // h_src[0..15]
    v16h hf1 = *(const v16h*)(hp + 16);  // h_src[16..31]
    v8h  eaA = *(const v8h*)(ep + 16 * hb);        // ea[j], j in lane's K-slice
    v8h  eaB = *(const v8h*)(ep + 16 * hb + 8);

    if (tile + wavesPerGrid < ntiles)  // -> global_prefetch_b8
      __builtin_prefetch(ea16 + (size_t)(tile + wavesPerGrid) * 16 * H, 0, 3);

    v8f acc0 = {}, acc1 = {};
    v16h abuf[2];  // double-buffered A fragment: avoids WMMA->VALU WAR nops

#pragma unroll
    for (int i = 0; i < 32; ++i) {  // K-steps over the outer-product part
      _Float16 hv = (i < 16) ? hf0[i] : hf1[i - 16];
      v8h hs;  // splat -> vector muls lower to v_pk_mul_f16
#pragma unroll
      for (int c = 0; c < 8; ++c) hs[c] = hv;
      abuf[i & 1] = cat8(hs * eaA, hs * eaB);
      v16h b0 = cat8(*(const v8h*)(brow0 + 32 * i), *(const v8h*)(brow0 + 32 * i + 8));
      v16h b1 = cat8(*(const v8h*)(brow1 + 32 * i), *(const v8h*)(brow1 + 32 * i + 8));
      acc0 = __builtin_amdgcn_wmma_f32_16x16x32_f16(false, abuf[i & 1], false, b0, (short)0, acc0, false, false);
      acc1 = __builtin_amdgcn_wmma_f32_16x16x32_f16(false, abuf[i & 1], false, b1, (short)0, acc1, false, false);
    }
    {  // bias K-step: A columns 1024..1055 = h_src itself
      v8h hA = *(const v8h*)(hp + 16 * hb);
      v8h hB = *(const v8h*)(hp + 16 * hb + 8);
      v16h a = cat8(hA, hB);
      v16h b0 = cat8(*(const v8h*)(brow0 + 32 * 32), *(const v8h*)(brow0 + 32 * 32 + 8));
      v16h b1 = cat8(*(const v8h*)(brow1 + 32 * 32), *(const v8h*)(brow1 + 32 * 32 + 8));
      acc0 = __builtin_amdgcn_wmma_f32_16x16x32_f16(false, a, false, b0, (short)0, acc0, false, false);
      acc1 = __builtin_amdgcn_wmma_f32_16x16x32_f16(false, a, false, b1, (short)0, acc1, false, false);
    }

    // C layout: lane(lo,hb), vgpr r -> M = 8*hb + r, N = lo (+16 for acc1)
#pragma unroll
    for (int r = 0; r < 8; ++r) {
      int m = 8 * hb + r;
      int d = dstIdx[tile * 16 + m];
      atomicAdd(&agg[(size_t)d * H + lo], acc0[r]);
      atomicAdd(&agg[(size_t)d * H + 16 + lo], acc1[r]);
    }
  }
}

// ---- s = relu(agg + hin @ Wroot[k] + bconv[k]) + hin ; also f16 copy ----
__global__ void node_update_kernel(const float* __restrict__ agg, const float* __restrict__ hin,
                                   const float* __restrict__ Wroot, const float* __restrict__ bconv,
                                   int k, float* __restrict__ sout, _Float16* __restrict__ s16) {
  int t = blockIdx.x * blockDim.x + threadIdx.x;
  if (t >= NN * H) return;
  int n = t >> 5, o = t & 31;
  float acc = agg[t] + bconv[k * H + o];
  const float* hi = hin + (size_t)n * H;
#pragma unroll
  for (int i = 0; i < H; ++i) acc += hi[i] * Wroot[(size_t)k * H * H + i * H + o];
  float s = fmaxf(acc, 0.f) + hi[o];
  sout[t] = s;
  s16[t] = (_Float16)s;
}

__global__ void pool_kernel(const float* __restrict__ s3, const int* __restrict__ batch,
                            float* __restrict__ pooled) {
  int t = blockIdx.x * blockDim.x + threadIdx.x;
  if (t >= NN * H) return;
  int n = t >> 5, o = t & 31;
  atomicAdd(&pooled[(size_t)batch[n] * H + o], s3[t]);
}

__global__ void mlp1_kernel(const float* __restrict__ pooled, const float* __restrict__ W1,
                            const float* __restrict__ b1, float* __restrict__ m1) {
  int t = blockIdx.x * blockDim.x + threadIdx.x;
  if (t >= GG * 128) return;
  int g = t >> 7, c = t & 127;
  float acc = b1[c];
#pragma unroll
  for (int i = 0; i < H; ++i) acc += pooled[g * H + i] * W1[i * 128 + c];
  m1[t] = fmaxf(acc, 0.f);
}

__global__ void mlp2_kernel(const float* __restrict__ m1, const float* __restrict__ W2,
                            const float* __restrict__ b2, float* __restrict__ m2) {
  int t = blockIdx.x * blockDim.x + threadIdx.x;
  if (t >= GG * 64) return;
  int g = t >> 6, c = t & 63;
  float acc = b2[c];
#pragma unroll
  for (int i = 0; i < 128; ++i) acc += m1[g * 128 + i] * W2[i * 64 + c];
  m2[t] = fmaxf(acc, 0.f);
}

__global__ void mlp3_kernel(const float* __restrict__ m2, const float* __restrict__ W3,
                            const float* __restrict__ b3, float* __restrict__ out) {
  int g = blockIdx.x * blockDim.x + threadIdx.x;
  if (g >= GG) return;
  float acc = b3[0];
#pragma unroll
  for (int i = 0; i < 64; ++i) acc += m2[g * 64 + i] * W3[i];
  out[g] = acc;
}

extern "C" void kernel_launch(void* const* d_in, const int* in_sizes, int n_in,
                              void* d_out, int out_size, void* d_ws, size_t ws_size,
                              hipStream_t stream) {
  const float* x     = (const float*)d_in[0];
  const int*   ei    = (const int*)d_in[1];     // [2, E]
  const float* eattr = (const float*)d_in[2];
  const int*   batch = (const int*)d_in[3];
  const float* Wn    = (const float*)d_in[4];
  const float* bn    = (const float*)d_in[5];
  const float* We    = (const float*)d_in[6];
  const float* be    = (const float*)d_in[7];
  const float* Wnn   = (const float*)d_in[8];
  const float* bnn   = (const float*)d_in[9];
  const float* Wroot = (const float*)d_in[10];
  const float* bconv = (const float*)d_in[11];
  const float* W1    = (const float*)d_in[12];
  const float* b1    = (const float*)d_in[13];
  const float* W2    = (const float*)d_in[14];
  const float* b2    = (const float*)d_in[15];
  const float* W3    = (const float*)d_in[16];
  const float* b3    = (const float*)d_in[17];
  float* out = (float*)d_out;

  const int* srcIdx = ei;
  const int* dstIdx = ei + NE;

  // bump allocator over workspace
  char* wsp = (char*)d_ws;
  auto alloc = [&](size_t bytes) -> char* {
    char* p = wsp;
    wsp += (bytes + 255) & ~(size_t)255;
    return p;
  };
  float*    h0   = (float*)alloc((size_t)NN * H * 4);
  float*    h1   = (float*)alloc((size_t)NN * H * 4);
  _Float16* h160 = (_Float16*)alloc((size_t)NN * H * 2);
  _Float16* h161 = (_Float16*)alloc((size_t)NN * H * 2);
  _Float16* ea16 = (_Float16*)alloc((size_t)NE * H * 2);
  _Float16* Bp   = (_Float16*)alloc((size_t)H * KTOT * 2);
  float*    agg  = (float*)alloc((size_t)NN * H * 4);
  float*    pooled = (float*)alloc((size_t)GG * H * 4);
  float*    m1   = (float*)alloc((size_t)GG * 128 * 4);
  float*    m2   = (float*)alloc((size_t)GG * 64 * 4);

  const int T = 256;
  const int gNH = (NN * H + T - 1) / T;
  const int ntiles = NE / 16;  // 7500 exactly

  node_embed_kernel<<<gNH, T, 0, stream>>>(x, Wn, bn, h0, h160);
  edge_embed_kernel<<<(NE * H + T - 1) / T, T, 0, stream>>>(eattr, We, be, ea16);

  float* hin = h0;  _Float16* hin16 = h160;
  float* hout = h1; _Float16* hout16 = h161;
  const size_t shBytes = (size_t)H * BROW * 2;  // 68096 B of LDS for B'
  for (int k = 0; k < 3; ++k) {
    zero_kernel<<<gNH, T, 0, stream>>>(agg, NN * H);
    bprep_kernel<<<(H * KTOT + T - 1) / T, T, 0, stream>>>(Wnn, bnn, Bp, k);
    edge_msg_kernel<<<256, T, shBytes, stream>>>(hin16, ea16, Bp, srcIdx, dstIdx, agg, ntiles);
    node_update_kernel<<<gNH, T, 0, stream>>>(agg, hin, Wroot, bconv, k, hout, hout16);
    { float* tf = hin; hin = hout; hout = tf; }
    { _Float16* th = hin16; hin16 = hout16; hout16 = th; }
  }
  // hin now holds s3
  zero_kernel<<<(GG * H + T - 1) / T, T, 0, stream>>>(pooled, GG * H);
  pool_kernel<<<gNH, T, 0, stream>>>(hin, batch, pooled);
  mlp1_kernel<<<(GG * 128 + T - 1) / T, T, 0, stream>>>(pooled, W1, b1, m1);
  mlp2_kernel<<<(GG * 64 + T - 1) / T, T, 0, stream>>>(m1, W2, b2, m2);
  mlp3_kernel<<<(GG + T - 1) / T, T, 0, stream>>>(m2, W3, b3, out);
}
